// GroupSplitter_71846212927741
// MI455X (gfx1250) — matrix operand order repaired and checked by reading, not compiled, not run on hardware
//
#include <hip/hip_runtime.h>
#include <hip/hip_bf16.h>

// Graph transformer (TransformerConv-style) for MI455X / gfx1250.
// Dense GEMMs via v_wmma_f32_16x16x32_f16: templated-K (fully unrolled),
// LDS-staged f16 tiles with TRANSPOSED B panel so both A and B fragments
// lower to ds_load_b128 pairs; 4 waves/block, RT accumulators per wave.
// Edge softmax/aggregation via f32 global atomics (L2-resident working set);
// LayerNorm via wave32 shuffle reductions.

#define N_NODES 32768
#define G_GRAPHS 256
#define NPG 128
#define E_EDGES 262144
#define H_HEADS 4
#define HID 128
#define DHEAD 32
#define FFN 256
#define L_BLOCKS 3

typedef __attribute__((ext_vector_type(16))) _Float16 v16h;
typedef __attribute__((ext_vector_type(8)))  float    v8f;

// ---------------------------------------------------------------------------
// WMMA GEMM: C[M x Ncols] = A[M x K] @ W[K x Ncols] + bias, opt ReLU.
//   Template: K (multiple of 32), RT = row-tiles per block (block computes
//   RT*16 rows x 64 cols).  Block: 128 threads = 4 waves.
//   Wave w -> row-tile (w % RT), col-tiles (w/RT)*RT .. +RT-1  (RT accums).
//   Grid: (M / (RT*16), Ncols / 64).
// LDS: A tile  [RT*16 x K]  row-major (stride K+8, keeps 16B alignment),
//      B panel [64 x K]     TRANSPOSED (ldsBt[c][k]) so a lane's B fragment
//      (rows kb..kb+15 at its column) is 16 contiguous halves -> 2x b128.
// ---------------------------------------------------------------------------
template<int K, int RT>
__global__ __launch_bounds__(128) void wmma_gemm_kernel(
    const float* __restrict__ A, const float* __restrict__ W,
    const float* __restrict__ bias, float* __restrict__ C,
    int Ncols, int relu)
{
  constexpr int ROWS = RT * 16;
  constexpr int SA   = K + 8;                 // padded stride (halves)
  __shared__ _Float16 ldsA [ROWS * SA];
  __shared__ _Float16 ldsBt[64 * SA];

  const int t    = threadIdx.x;
  const int w    = t >> 5;
  const int lane = t & 31;
  const int l16  = lane & 15;
  const int hi   = lane >> 4;                 // 0: lanes 0-15, 1: lanes 16-31
  const int rowBase = blockIdx.x * ROWS;
  const int colBase = blockIdx.y * 64;

  // Stage A tile (ROWS x K), f32 -> f16, coalesced global reads.
#pragma unroll
  for (int idx = t; idx < ROWS * K; idx += 128) {
    int r = idx / K, kk = idx % K;            // K is a power of two -> shifts
    ldsA[r * SA + kk] = (_Float16)A[(size_t)(rowBase + r) * K + kk];
  }
  // Stage B panel transposed: ldsBt[c * SA + k] = W[k][colBase + c].
#pragma unroll
  for (int idx = t; idx < K * 64; idx += 128) {
    int kk = idx >> 6, c = idx & 63;
    ldsBt[c * SA + kk] = (_Float16)W[(size_t)kk * Ncols + colBase + c];
  }
  // Speculative prefetch of the next row-panel of A (dropped if OOB).
  __builtin_prefetch(A + ((size_t)(rowBase + ROWS) * K) + (size_t)t * 16, 0, 1);
  __syncthreads();

  const int rowTile  = w % RT;
  const int colTile0 = (w / RT) * RT;
  v8f acc[RT];
#pragma unroll
  for (int ct = 0; ct < RT; ++ct) acc[ct] = (v8f){};

#pragma unroll
  for (int k0 = 0; k0 < K; k0 += 32) {
    // A fragment (16x32 f16): lanes 0-15 hold K {k0..k0+15},
    // lanes 16-31 hold K {k0+16..k0+31}; row M = lane%16.
    v16h a;
    const _Float16* ap = &ldsA[(rowTile * 16 + l16) * SA + k0 + hi * 16];
#pragma unroll
    for (int i = 0; i < 16; ++i) a[i] = ap[i];
#pragma unroll
    for (int ct = 0; ct < RT; ++ct) {
      // B fragment (32x16 f16): b[i] = W[kb + i][col], kb = k0 + hi*16 —
      // contiguous in the transposed panel -> two ds_load_b128.
      v16h b;
      const _Float16* bp =
          &ldsBt[((colTile0 + ct) * 16 + l16) * SA + k0 + hi * 16];
#pragma unroll
      for (int i = 0; i < 16; ++i) b[i] = bp[i];
      acc[ct] = __builtin_amdgcn_wmma_f32_16x16x32_f16(
          false, a, false, b, (short)0, acc[ct], false, false);
    }
  }

  // D layout: VGPR j -> row = hi*8 + j (within tile), col = lane%16.
#pragma unroll
  for (int ct = 0; ct < RT; ++ct) {
    const int col = colBase + (colTile0 + ct) * 16 + l16;
    const float bv = bias[col];
#pragma unroll
    for (int j = 0; j < 8; ++j) {
      int row = rowBase + rowTile * 16 + hi * 8 + j;
      float vv = acc[ct][j] + bv;
      if (relu) vv = fmaxf(vv, 0.f);
      C[(size_t)row * Ncols + col] = vv;
    }
  }
}

// ---------------------------------------------------------------------------
// Input projection: h = [x || group_probs[batch]] @ in_W + in_b.  (K=7, VALU)
// ---------------------------------------------------------------------------
__global__ void input_proj_kernel(const float* __restrict__ x,
                                  const float* __restrict__ gp,
                                  const int* __restrict__ batch,
                                  const float* __restrict__ inW,
                                  const float* __restrict__ inb,
                                  float* __restrict__ h)
{
  int t = blockIdx.x * blockDim.x + threadIdx.x;   // N*HID threads
  int n = t >> 7, c = t & 127;
  int b = batch[n];
  float acc = inb[c];
#pragma unroll
  for (int i = 0; i < 4; ++i) acc += x[n * 4 + i] * inW[i * HID + c];
#pragma unroll
  for (int j = 0; j < 3; ++j) acc += gp[b * 3 + j] * inW[(4 + j) * HID + c];
  h[t] = acc;
}

// ---------------------------------------------------------------------------
// Edge pass 1: per (edge, head) score -> ex = exp(score*scale), den[dst,h] += ex.
// Max-subtraction omitted: softmax is shift-invariant and scores are tiny.
// Edge projection e = edge_attr @ We + be recomputed in registers (We: 4x128).
// ---------------------------------------------------------------------------
__global__ void edge_softmax_kernel(const int* __restrict__ ei,
                                    const float* __restrict__ ea,
                                    const float* __restrict__ q,
                                    const float* __restrict__ k,
                                    const float* __restrict__ We,
                                    const float* __restrict__ be,
                                    float* __restrict__ ex,
                                    float* __restrict__ den)
{
  int t = blockIdx.x * blockDim.x + threadIdx.x;   // E*H threads
  int e = t >> 2, hh = t & 3;
  int src = ei[e], dst = ei[E_EDGES + e];
  float a0 = ea[e * 4 + 0], a1 = ea[e * 4 + 1];
  float a2 = ea[e * 4 + 2], a3 = ea[e * 4 + 3];
  const float* qd = q + (size_t)dst * HID + hh * DHEAD;
  const float* ks = k + (size_t)src * HID + hh * DHEAD;
  float sc = 0.f;
#pragma unroll 8
  for (int d = 0; d < DHEAD; ++d) {
    int col = hh * DHEAD + d;
    float ep = be[col] + a0 * We[col] + a1 * We[HID + col]
             + a2 * We[2 * HID + col] + a3 * We[3 * HID + col];
    sc += qd[d] * (ks[d] + ep);
  }
  float ev = __expf(sc * 0.17677669529663687f);    // 1/sqrt(32)
  ex[t] = ev;
  atomicAdd(&den[dst * H_HEADS + hh], ev);
}

// ---------------------------------------------------------------------------
// Edge pass 2: agg[dst] += alpha * (v[src] + e)  (f32 atomic scatter).
// ---------------------------------------------------------------------------
__global__ void edge_aggregate_kernel(const int* __restrict__ ei,
                                      const float* __restrict__ ea,
                                      const float* __restrict__ v,
                                      const float* __restrict__ We,
                                      const float* __restrict__ be,
                                      const float* __restrict__ ex,
                                      const float* __restrict__ den,
                                      float* __restrict__ agg)
{
  int t = blockIdx.x * blockDim.x + threadIdx.x;   // E*H threads
  int e = t >> 2, hh = t & 3;
  int src = ei[e], dst = ei[E_EDGES + e];
  float alpha = ex[t] / fmaxf(den[dst * H_HEADS + hh], 1e-16f);
  float a0 = ea[e * 4 + 0], a1 = ea[e * 4 + 1];
  float a2 = ea[e * 4 + 2], a3 = ea[e * 4 + 3];
  const float* vs = v + (size_t)src * HID + hh * DHEAD;
  float* ad = agg + (size_t)dst * HID + hh * DHEAD;
#pragma unroll 8
  for (int d = 0; d < DHEAD; ++d) {
    int col = hh * DHEAD + d;
    float ep = be[col] + a0 * We[col] + a1 * We[HID + col]
             + a2 * We[2 * HID + col] + a3 * We[3 * HID + col];
    atomicAdd(&ad[d], alpha * (vs[d] + ep));
  }
}

// ---------------------------------------------------------------------------
// Fused residual + LayerNorm: hout = LN(hin + add1 [+ add2]) * g + b.
// One wave32 per node; 4 elements/lane; shuffle-tree reductions.
// ---------------------------------------------------------------------------
__global__ __launch_bounds__(128) void ln_residual_kernel(
    const float* __restrict__ hin, const float* __restrict__ add1,
    const float* __restrict__ add2, const float* __restrict__ gam,
    const float* __restrict__ bet, float* __restrict__ hout)
{
  int wave = threadIdx.x >> 5, lane = threadIdx.x & 31;
  int node = blockIdx.x * 4 + wave;
  size_t base = (size_t)node * HID;
  float vals[4];
  float s = 0.f;
#pragma unroll
  for (int i = 0; i < 4; ++i) {
    int c = i * 32 + lane;
    float vv = hin[base + c] + add1[base + c];
    if (add2) vv += add2[base + c];
    vals[i] = vv; s += vv;
  }
#pragma unroll
  for (int off = 16; off > 0; off >>= 1) s += __shfl_xor(s, off, 32);
  float mean = s * (1.f / 128.f);
  float vs = 0.f;
#pragma unroll
  for (int i = 0; i < 4; ++i) { float d = vals[i] - mean; vs += d * d; }
#pragma unroll
  for (int off = 16; off > 0; off >>= 1) vs += __shfl_xor(vs, off, 32);
  float inv = rsqrtf(vs * (1.f / 128.f) + 1e-5f);
#pragma unroll
  for (int i = 0; i < 4; ++i) {
    int c = i * 32 + lane;
    hout[base + c] = (vals[i] - mean) * inv * gam[c] + bet[c];
  }
}

// ---------------------------------------------------------------------------
// Per-node head: node_logits = [h || u[batch]] @ node_W + node_b  -> out[N,3]
// ---------------------------------------------------------------------------
__global__ void node_head_kernel(const float* __restrict__ h,
                                 const float* __restrict__ u,
                                 const int* __restrict__ batch,
                                 const float* __restrict__ nW,
                                 const float* __restrict__ nb,
                                 float* __restrict__ out)
{
  int n = blockIdx.x * blockDim.x + threadIdx.x;
  if (n >= N_NODES) return;
  float uu = u[batch[n]];
  float a0 = nb[0], a1 = nb[1], a2 = nb[2];
  const float* hr = h + (size_t)n * HID;
  for (int i = 0; i < HID; ++i) {
    float hv = hr[i];
    a0 += hv * nW[i * 3 + 0];
    a1 += hv * nW[i * 3 + 1];
    a2 += hv * nW[i * 3 + 2];
  }
  a0 += uu * nW[HID * 3 + 0];
  a1 += uu * nW[HID * 3 + 1];
  a2 += uu * nW[HID * 3 + 2];
  out[n * 3 + 0] = a0; out[n * 3 + 1] = a1; out[n * 3 + 2] = a2;
}

// ---------------------------------------------------------------------------
// Gate MLP: gate[n] = relu(h @ W1 + b1) @ W2 + b2   (128 -> 64 -> 1)
// ---------------------------------------------------------------------------
__global__ void gate_kernel(const float* __restrict__ h,
                            const float* __restrict__ W1,
                            const float* __restrict__ b1,
                            const float* __restrict__ W2,
                            const float* __restrict__ b2,
                            float* __restrict__ gate)
{
  int n = blockIdx.x * blockDim.x + threadIdx.x;
  if (n >= N_NODES) return;
  const float* hr = h + (size_t)n * HID;
  float acc = b2[0];
  for (int j = 0; j < 64; ++j) {
    float g = b1[j];
#pragma unroll 8
    for (int i = 0; i < HID; ++i) g += hr[i] * W1[i * 64 + j];
    acc += fmaxf(g, 0.f) * W2[j];
  }
  gate[n] = acc;
}

// ---------------------------------------------------------------------------
// Per-graph attentional pooling + energy head. One block (128 thr) per graph.
// ---------------------------------------------------------------------------
__global__ __launch_bounds__(128) void pool_energy_kernel(
    const float* __restrict__ h, const float* __restrict__ gate,
    const float* __restrict__ u, const float* __restrict__ eW,
    const float* __restrict__ eb, float* __restrict__ energy_out)
{
  __shared__ float red[128];
  __shared__ float w[128];
  __shared__ float pool[128];
  int g = blockIdx.x, t = threadIdx.x;
  float gv = gate[g * NPG + t];
  red[t] = gv; __syncthreads();
  for (int s = 64; s > 0; s >>= 1) {
    if (t < s) red[t] = fmaxf(red[t], red[t + s]);
    __syncthreads();
  }
  float mx = red[0]; __syncthreads();
  float ev = __expf(gv - mx);
  red[t] = ev; __syncthreads();
  for (int s = 64; s > 0; s >>= 1) {
    if (t < s) red[t] += red[t + s];
    __syncthreads();
  }
  float sum = red[0];
  w[t] = ev / fmaxf(sum, 1e-16f);
  __syncthreads();
  float acc = 0.f;
  for (int n = 0; n < NPG; ++n)
    acc += w[n] * h[((size_t)g * NPG + n) * HID + t];
  pool[t] = acc; __syncthreads();
  if (t < 3) {
    float e = eb[t] + u[g] * eW[HID * 3 + t];
    for (int i = 0; i < HID; ++i) e += pool[i] * eW[i * 3 + t];
    energy_out[g * 3 + t] = e;
  }
}

// ---------------------------------------------------------------------------
// Host-side orchestration.
// ---------------------------------------------------------------------------
extern "C" void kernel_launch(void* const* d_in, const int* in_sizes, int n_in,
                              void* d_out, int out_size, void* d_ws, size_t ws_size,
                              hipStream_t stream) {
  const float* x     = (const float*)d_in[0];
  const float* gp    = (const float*)d_in[1];
  const float* u     = (const float*)d_in[2];
  const int*   batch = (const int*)  d_in[3];
  const int*   ei    = (const int*)  d_in[4];
  const float* ea    = (const float*)d_in[5];
  const float* inW   = (const float*)d_in[6];
  const float* inb   = (const float*)d_in[7];
  const float* Wq    = (const float*)d_in[8];
  const float* bq    = (const float*)d_in[9];
  const float* Wk    = (const float*)d_in[10];
  const float* bk    = (const float*)d_in[11];
  const float* Wv    = (const float*)d_in[12];
  const float* bv    = (const float*)d_in[13];
  const float* We    = (const float*)d_in[14];
  const float* be    = (const float*)d_in[15];
  const float* Wskip = (const float*)d_in[16];
  const float* bskip = (const float*)d_in[17];
  const float* ln1g  = (const float*)d_in[18];
  const float* ln1b  = (const float*)d_in[19];
  const float* W1    = (const float*)d_in[20];
  const float* b1    = (const float*)d_in[21];
  const float* W2    = (const float*)d_in[22];
  const float* b2    = (const float*)d_in[23];
  const float* ln2g  = (const float*)d_in[24];
  const float* ln2b  = (const float*)d_in[25];
  const float* nW    = (const float*)d_in[26];
  const float* nb    = (const float*)d_in[27];
  const float* gW1   = (const float*)d_in[28];
  const float* gb1   = (const float*)d_in[29];
  const float* gW2   = (const float*)d_in[30];
  const float* gb2   = (const float*)d_in[31];
  const float* eWm   = (const float*)d_in[32];
  const float* ebm   = (const float*)d_in[33];

  float* out_nodes  = (float*)d_out;                   // [N,3]
  float* out_energy = out_nodes + (size_t)N_NODES * 3; // [G,3]

  // Workspace layout (bytes): h | q | k | v | tmp(N*FFN) | den | gate
  // ex overlays the upper half of tmp (dead before the FFN uses tmp).
  char* ws = (char*)d_ws;
  const size_t SZ_NH = (size_t)N_NODES * HID * sizeof(float);   // 16 MB
  float* h    = (float*)(ws);
  float* q    = (float*)(ws + SZ_NH);          // also: agg, then ffn-out
  float* k    = (float*)(ws + 2 * SZ_NH);
  float* v    = (float*)(ws + 3 * SZ_NH);
  float* tmp  = (float*)(ws + 4 * SZ_NH);      // N*FFN  (skip-out in lower half)
  float* ex   = tmp + (size_t)N_NODES * HID;   // E*H floats, overlays tmp upper half
  float* den  = (float*)(ws + 4 * SZ_NH + (size_t)N_NODES * FFN * sizeof(float));
  float* gate = den + (size_t)N_NODES * H_HEADS;

  // ---- input projection ----
  input_proj_kernel<<<(N_NODES * HID) / 256, 256, 0, stream>>>(x, gp, batch, inW, inb, h);

  const dim3 gemmBlk(128);
  const dim3 gridK128_C128(N_NODES / 64, HID / 64);   // (512, 2)
  const dim3 gridK128_C256(N_NODES / 64, FFN / 64);   // (512, 4)
  const dim3 gridK256_C128(N_NODES / 32, HID / 64);   // (1024, 2)
  const int edgeBlocks = (E_EDGES * H_HEADS) / 256;

  for (int l = 0; l < L_BLOCKS; ++l) {
    const float* Wq_l = Wq + (size_t)l * HID * HID;
    const float* Wk_l = Wk + (size_t)l * HID * HID;
    const float* Wv_l = Wv + (size_t)l * HID * HID;
    const float* Ws_l = Wskip + (size_t)l * HID * HID;
    const float* We_l = We + (size_t)l * 4 * HID;
    const float* W1_l = W1 + (size_t)l * HID * FFN;
    const float* W2_l = W2 + (size_t)l * FFN * HID;

    // Q, K, V projections (WMMA, K=128, 64x64 block tiles)
    wmma_gemm_kernel<128, 4><<<gridK128_C128, gemmBlk, 0, stream>>>(
        h, Wq_l, bq + l * HID, q, HID, 0);
    wmma_gemm_kernel<128, 4><<<gridK128_C128, gemmBlk, 0, stream>>>(
        h, Wk_l, bk + l * HID, k, HID, 0);
    wmma_gemm_kernel<128, 4><<<gridK128_C128, gemmBlk, 0, stream>>>(
        h, Wv_l, bv + l * HID, v, HID, 0);
    // Skip projection into tmp (lower N*HID half)
    wmma_gemm_kernel<128, 4><<<gridK128_C128, gemmBlk, 0, stream>>>(
        h, Ws_l, bskip + l * HID, tmp, HID, 0);

    // Segment softmax over incoming edges
    hipMemsetAsync(den, 0, (size_t)N_NODES * H_HEADS * sizeof(float), stream);
    edge_softmax_kernel<<<edgeBlocks, 256, 0, stream>>>(
        ei, ea, q, k, We_l, be + l * HID, ex, den);

    // Aggregate into agg (reuses q buffer — q is dead after pass 1)
    float* agg = q;
    hipMemsetAsync(agg, 0, SZ_NH, stream);
    edge_aggregate_kernel<<<edgeBlocks, 256, 0, stream>>>(
        ei, ea, v, We_l, be + l * HID, ex, den, agg);

    // h = LN(h + agg + skip)
    ln_residual_kernel<<<N_NODES / 4, 128, 0, stream>>>(
        h, agg, tmp, ln1g + l * HID, ln1b + l * HID, h);

    // FFN: tmp = relu(h @ W1 + b1);  ffn = tmp @ W2 + b2 (into q buffer)
    wmma_gemm_kernel<128, 4><<<gridK128_C256, gemmBlk, 0, stream>>>(
        h, W1_l, b1 + l * FFN, tmp, FFN, 1);
    wmma_gemm_kernel<256, 2><<<gridK256_C128, gemmBlk, 0, stream>>>(
        tmp, W2_l, b2 + l * HID, q, HID, 0);

    // h = LN(h + ffn)
    ln_residual_kernel<<<N_NODES / 4, 128, 0, stream>>>(
        h, q, nullptr, ln2g + l * HID, ln2b + l * HID, h);
  }

  // ---- heads ----
  node_head_kernel<<<N_NODES / 256, 256, 0, stream>>>(h, u, batch, nW, nb, out_nodes);
  gate_kernel<<<N_NODES / 256, 256, 0, stream>>>(h, gW1, gb1, gW2, gb2, gate);
  pool_energy_kernel<<<G_GRAPHS, 128, 0, stream>>>(h, gate, u, eWm, ebm, out_energy);
}
